// MFN_35794257445298
// MI455X (gfx1250) — compile-verified, matrix operand order checked by reading, not run on hardware
//
#include <hip/hip_runtime.h>
#include <hip/hip_bf16.h>
#include <cstddef>

// ---------------------------------------------------------------------------
// MFN recurrent network on gfx1250 (MI455X).
// Persistent kernel: 32 blocks x 256 threads (8 wave32). Each block owns a
// 16-row batch tile and runs all 128 timesteps with state in LDS.
// All GEMMs via v_wmma_f32_16x16x32_f16 (f16 in, fp32 accumulate).
// ---------------------------------------------------------------------------

typedef __attribute__((ext_vector_type(16))) _Float16 v16h;
typedef __attribute__((ext_vector_type(8)))  float    v8f;

// Problem dims
constexpr int SEQ = 128, BATCH = 512, XC = 752;
constexpr int T_IN = 300, A_IN = 81, V_IN = 371;

// Fragment = 32 lanes x 16 halfs = 512 halfs (1 KB)
constexpr size_t FR = 512;

// Weight-fragment arena offsets in d_ws (units: halfs).
// Layout per matrix: [coltile][ktile][lane][16]
constexpr size_t OFF_TCAT = 0;                          // [Wih(300->320)|Whh(128)]: 32ct x 14kt
constexpr size_t OFF_ACAT = OFF_TCAT + 32 * 14 * FR;    // [Wih(81->96)|Whh(32)]:    8ct x 4kt
constexpr size_t OFF_VCAT = OFF_ACAT + 8 * 4 * FR;      // [Wih(371->384)|Whh(32)]:  8ct x 13kt
constexpr size_t OFF_A1W1 = OFF_VCAT + 8 * 13 * FR;     // attn1_w1 [128,384]:       8ct x 12kt
constexpr size_t OFF_A1W2 = OFF_A1W1 + 8 * 12 * FR;     // attn1_w2 [384,128]:      24ct x 4kt
constexpr size_t OFF_A2W1 = OFF_A1W2 + 24 * 4 * FR;     // attn2_w1 [128,384]:       8ct x 12kt
constexpr size_t OFF_A2W2 = OFF_A2W1 + 8 * 12 * FR;     // attn2_w2 [128,128]:       8ct x 4kt
constexpr size_t OFF_G1W1 = OFF_A2W2 + 8 * 4 * FR;      // g1_w1 [128,512]:          8ct x 16kt
constexpr size_t OFF_G1W2 = OFF_G1W1 + 8 * 16 * FR;     // g1_w2 [128,128]
constexpr size_t OFF_G2W1 = OFF_G1W2 + 8 * 4 * FR;      // g2_w1 [128,512]
constexpr size_t OFF_G2W2 = OFF_G2W1 + 8 * 16 * FR;     // g2_w2 [128,128]
constexpr size_t OFF_OW1  = OFF_G2W2 + 8 * 4 * FR;      // out_w1 [128,320]:         8ct x 10kt
constexpr size_t WS_HALFS = OFF_OW1 + 8 * 10 * FR;      // ~1.3 MB total

// ---------------------------------------------------------------------------
// One-time weight prep: fp32 -> f16, swizzled into the exact WMMA B-fragment
// layout (ISA 7.12.2 16-bit operand layout). Supports K-concatenated [W1|W2]
// so each LSTM runs as a single fused GEMM over [x, h].
// ---------------------------------------------------------------------------
__global__ void prep_weight_frags(const float* __restrict__ W1, int K1, int K1pad,
                                  const float* __restrict__ W2, int K2,
                                  int N, _Float16* __restrict__ dst) {
  const int ktiles = (K1pad + K2) >> 5;
  const int total = (N >> 4) * ktiles * 32;
  const int tid = blockIdx.x * blockDim.x + threadIdx.x;
  if (tid >= total) return;
  const int lane = tid & 31;
  const int frag = tid >> 5;
  const int ct = frag / ktiles, kt = frag - ct * ktiles;
  const int n = ct * 16 + (lane & 15);
  const int g = lane >> 4;
  _Float16* o = dst + (size_t)frag * FR + lane * 16;
#pragma unroll
  for (int e = 0; e < 16; ++e) {
    const int v = e >> 1, p = e & 1;
    const int k = kt * 32 + ((v & 3) << 1) + p + (g << 3) + ((e >= 8) ? 16 : 0);
    float val = 0.f;
    if (k < K1)                           val = W1[(size_t)n * K1 + k];
    else if (k >= K1pad && (k - K1pad) < K2) val = W2[(size_t)n * K2 + (k - K1pad)];
    o[e] = (_Float16)val;
  }
}

// ---------------------------------------------------------------------------
// Device helpers
// ---------------------------------------------------------------------------
__device__ __forceinline__ float sigm(float x) { return 1.f / (1.f + __expf(-x)); }

// Stage one 16x32 f16 A-fragment from an fp32 row-major source (LDS or global).
__device__ __forceinline__ void stage_frag(const float* __restrict__ src, int ld, int K,
                                           int k0, _Float16* __restrict__ dst, int lane) {
  const int m = lane & 15, g = lane >> 4;
  _Float16* o = dst + lane * 16;
#pragma unroll
  for (int e = 0; e < 16; ++e) {
    const int v = e >> 1, p = e & 1;
    const int k = k0 + ((v & 3) << 1) + p + (g << 3) + ((e >= 8) ? 16 : 0);
    float f = 0.f;
    if (k < K) f = src[(size_t)m * ld + k];
    o[e] = (_Float16)f;
  }
}

// One 16-column output tile: D[16x16] = A[16xK] x W^T tile + bias, activation.
// act: 0=none 1=relu 2=sigmoid 3=tanh
__device__ __forceinline__ void gemm_ct(const _Float16* __restrict__ afr,
                                        const _Float16* __restrict__ wfr, int nkt,
                                        int ct, int lane,
                                        float* __restrict__ dst, int ldo,
                                        const float* __restrict__ biasA,
                                        const float* __restrict__ biasB, int act) {
  v8f acc = {};
  const _Float16* wp = wfr + (size_t)ct * nkt * FR;
  for (int kt = 0; kt < nkt; ++kt) {
    v16h a = *(const v16h*)(afr + (size_t)kt * FR + lane * 16);
    v16h b = *(const v16h*)(wp + (size_t)kt * FR + lane * 16);
    acc = __builtin_amdgcn_wmma_f32_16x16x32_f16(false, a, false, b, (short)0, acc,
                                                 false, false);
  }
  const int n = ct * 16 + (lane & 15);
  const int g = lane >> 4;
  float bias = 0.f;
  if (biasA) bias += biasA[n];
  if (biasB) bias += biasB[n];
#pragma unroll
  for (int e = 0; e < 8; ++e) {
    float x = acc[e] + bias;
    if (act == 1)      x = fmaxf(x, 0.f);
    else if (act == 2) x = sigm(x);
    else if (act == 3) x = tanhf(x);
    dst[(size_t)(e + (g << 3)) * ldo + n] = x;
  }
}

// ---------------------------------------------------------------------------
// Persistent recurrent kernel. blockDim = 256 (8 wave32), gridDim = 32.
// Dynamic LDS: 192 KB per workgroup (CDNA5 WGP has 320 KB).
// ---------------------------------------------------------------------------
__global__ __launch_bounds__(256) void mfn_rnn_kernel(
    const float* __restrict__ x_p,
    const float* __restrict__ c_t0, const float* __restrict__ c_a0,
    const float* __restrict__ c_v0, const float* __restrict__ mem0,
    const float* __restrict__ t_bih, const float* __restrict__ t_bhh,
    const float* __restrict__ a_bih, const float* __restrict__ a_bhh,
    const float* __restrict__ v_bih, const float* __restrict__ v_bhh,
    const float* __restrict__ a1b1, const float* __restrict__ a1b2,
    const float* __restrict__ a2b1, const float* __restrict__ a2b2,
    const float* __restrict__ g1b1, const float* __restrict__ g1b2,
    const float* __restrict__ g2b1, const float* __restrict__ g2b2,
    const float* __restrict__ ob1, const float* __restrict__ ow2,
    const float* __restrict__ ob2,
    const _Float16* __restrict__ wf,
    float* __restrict__ out) {
  extern __shared__ __align__(16) char smem[];
  _Float16* pool = (_Float16*)smem;               // 32 frags       32768 B
  float* th   = (float*)(smem + 32768);           // [16][128]       8192
  float* tc   = (float*)(smem + 40960);           // [16][128]       8192
  float* ah   = (float*)(smem + 49152);           // [16][32]        2048
  float* ac   = (float*)(smem + 51200);           // [16][32]        2048
  float* vh   = (float*)(smem + 53248);           // [16][32]        2048
  float* vc   = (float*)(smem + 55296);           // [16][32]        2048
  float* mst  = (float*)(smem + 57344);           // [16][128]       8192
  float* cstar= (float*)(smem + 65536);           // [16][384]      24576
  float* gt   = (float*)(smem + 90112);           // [16][512]      32768
  float* ga   = (float*)(smem + 122880);          // [16][128]       8192
  float* gv   = (float*)(smem + 131072);          // [16][128]       8192
  float* Lb   = (float*)(smem + 139264);          // [16][384]      24576
  float* h1   = (float*)(smem + 163840);          // [16][128]       8192
  float* chat = (float*)(smem + 172032);          // [16][128]       8192
  float* g1v  = (float*)(smem + 180224);          // [16][128]       8192
  float* g2v  = (float*)(smem + 188416);          // [16][128]       8192
                                                  // total 196608 B

  const int tid = threadIdx.x;
  const int lane = tid & 31;
  const int wave = tid >> 5;
  const int b0 = blockIdx.x * 16;

  // --- init carry: h = 0, c = inputs, mem = input ---
  for (int idx = tid; idx < 16 * 128; idx += 256) {
    const int r = idx >> 7, j = idx & 127;
    th[idx] = 0.f;
    tc[idx] = c_t0[(size_t)(b0 + r) * 128 + j];
    mst[idx] = mem0[(size_t)(b0 + r) * 128 + j];
  }
  for (int idx = tid; idx < 16 * 32; idx += 256) {
    const int r = idx >> 5, j = idx & 31;
    ah[idx] = 0.f; ac[idx] = c_a0[(size_t)(b0 + r) * 32 + j];
    vh[idx] = 0.f; vc[idx] = c_v0[(size_t)(b0 + r) * 32 + j];
  }
  __syncthreads();

  for (int s = 0; s < SEQ; ++s) {
    const float* xr = x_p + ((size_t)s * BATCH + b0) * XC;

    // S1: stage LSTM A-fragments ([x|h] per modality), frag pool layout:
    //   Gt: [0..13] = xt(10) + th(4); Ga: [14..17] = xa(3)+ah; Gv: [18..30] = xv(12)+vh
    for (int j = wave; j < 31; j += 8) {
      if (j < 10)       stage_frag(xr,       XC, T_IN, j * 32,        pool + (size_t)j * FR, lane);
      else if (j < 14)  stage_frag(th,       128, 128, (j - 10) * 32, pool + (size_t)j * FR, lane);
      else if (j < 17)  stage_frag(xr + 300, XC, A_IN, (j - 14) * 32, pool + (size_t)j * FR, lane);
      else if (j == 17) stage_frag(ah,        32,  32, 0,             pool + (size_t)j * FR, lane);
      else if (j < 30)  stage_frag(xr + 381, XC, V_IN, (j - 18) * 32, pool + (size_t)j * FR, lane);
      else              stage_frag(vh,        32,  32, 0,             pool + (size_t)j * FR, lane);
    }
    // prefetch next step's x tile (gfx1250 global_prefetch_b8), 128B lines
    if (s + 1 < SEQ) {
      const float* nx = x_p + ((size_t)(s + 1) * BATCH + b0) * XC;
      for (int l = tid; l < 376; l += 256) __builtin_prefetch(nx + l * 32, 0, 1);
    }
    __syncthreads();

    // S2: fused LSTM gate GEMMs, G = [x,h] @ [Wih|Whh]^T + bih + bhh
    for (int t = wave; t < 48; t += 8) {
      if (t < 32)      gemm_ct(pool,                 wf + OFF_TCAT, 14, t,      lane, gt, 512, t_bih, t_bhh, 0);
      else if (t < 40) gemm_ct(pool + 14 * FR,       wf + OFF_ACAT,  4, t - 32, lane, ga, 128, a_bih, a_bhh, 0);
      else             gemm_ct(pool + 18 * FR,       wf + OFF_VCAT, 13, t - 40, lane, gv, 128, v_bih, v_bhh, 0);
    }
    __syncthreads();

    // S3: LSTM elementwise update; capture pre_c / cur_c into c_star[16][384]
    for (int idx = tid; idx < 16 * 128; idx += 256) {
      const int r = idx >> 7, j = idx & 127;
      const float gi = gt[r * 512 + j],      gf = gt[r * 512 + 128 + j];
      const float gg = gt[r * 512 + 256 + j], go = gt[r * 512 + 384 + j];
      const float cold = tc[idx];
      cstar[r * 384 + j] = cold;                              // pre_c (t)
      const float c2 = sigm(gf) * cold + sigm(gi) * tanhf(gg);
      tc[idx] = c2;
      th[idx] = sigm(go) * tanhf(c2);
      cstar[r * 384 + 192 + j] = c2;                          // cur_c (t)
    }
    for (int idx = tid; idx < 16 * 32; idx += 256) {
      const int r = idx >> 5, j = idx & 31;
      {
        const float gi = ga[r * 128 + j],      gf = ga[r * 128 + 32 + j];
        const float gg = ga[r * 128 + 64 + j], go = ga[r * 128 + 96 + j];
        const float cold = ac[idx];
        cstar[r * 384 + 128 + j] = cold;
        const float c2 = sigm(gf) * cold + sigm(gi) * tanhf(gg);
        ac[idx] = c2; ah[idx] = sigm(go) * tanhf(c2);
        cstar[r * 384 + 320 + j] = c2;
      }
      {
        const float gi = gv[r * 128 + j],      gf = gv[r * 128 + 32 + j];
        const float gg = gv[r * 128 + 64 + j], go = gv[r * 128 + 96 + j];
        const float cold = vc[idx];
        cstar[r * 384 + 160 + j] = cold;
        const float c2 = sigm(gf) * cold + sigm(gi) * tanhf(gg);
        vc[idx] = c2; vh[idx] = sigm(go) * tanhf(c2);
        cstar[r * 384 + 352 + j] = c2;
      }
    }
    __syncthreads();

    // S4: stage c_star frags [0..11]
    for (int j = wave; j < 12; j += 8) stage_frag(cstar, 384, 384, j * 32, pool + (size_t)j * FR, lane);
    __syncthreads();

    // S5: h1 = relu(c_star @ attn1_w1^T + b1)
    for (int t = wave; t < 8; t += 8) gemm_ct(pool, wf + OFF_A1W1, 12, t, lane, h1, 128, a1b1, nullptr, 1);
    __syncthreads();

    // S6: stage h1 frags [12..15]
    for (int j = wave; j < 4; j += 8) stage_frag(h1, 128, 128, j * 32, pool + (size_t)(12 + j) * FR, lane);
    __syncthreads();

    // S7: logits = h1 @ attn1_w2^T + b2  -> Lb[16][384]
    for (int t = wave; t < 24; t += 8) gemm_ct(pool + 12 * FR, wf + OFF_A1W2, 4, t, lane, Lb, 384, a1b2, nullptr, 0);
    __syncthreads();

    // S8: per-row softmax over 384, then attended = softmax * c_star (in Lb)
    for (int r = wave; r < 16; r += 8) {
      float mx = -3.0e38f;
      for (int c = lane; c < 384; c += 32) mx = fmaxf(mx, Lb[r * 384 + c]);
      for (int off = 16; off; off >>= 1) mx = fmaxf(mx, __shfl_xor(mx, off, 32));
      float sum = 0.f;
      for (int c = lane; c < 384; c += 32) {
        const float e = __expf(Lb[r * 384 + c] - mx);
        Lb[r * 384 + c] = e; sum += e;
      }
      for (int off = 16; off; off >>= 1) sum += __shfl_xor(sum, off, 32);
      const float inv = 1.f / sum;
      for (int c = lane; c < 384; c += 32) Lb[r * 384 + c] = Lb[r * 384 + c] * inv * cstar[r * 384 + c];
    }
    __syncthreads();

    // S9: stage "both" = [attended(12 frags) | mem(4 frags)] at pool[0..15]
    for (int j = wave; j < 16; j += 8) {
      if (j < 12) stage_frag(Lb, 384, 384, j * 32, pool + (size_t)j * FR, lane);
      else        stage_frag(mst, 128, 128, (j - 12) * 32, pool + (size_t)j * FR, lane);
    }
    __syncthreads();

    // S10: attn2 hidden (reuse h1), g1/g2 hiddens (reuse ga/gv), all relu
    for (int t = wave; t < 24; t += 8) {
      if (t < 8)       gemm_ct(pool, wf + OFF_A2W1, 12, t,      lane, h1, 128, a2b1, nullptr, 1);
      else if (t < 16) gemm_ct(pool, wf + OFF_G1W1, 16, t - 8,  lane, ga, 128, g1b1, nullptr, 1);
      else             gemm_ct(pool, wf + OFF_G2W1, 16, t - 16, lane, gv, 128, g2b1, nullptr, 1);
    }
    __syncthreads();

    // S11: stage hidden frags: h2 -> [16..19], hg1 -> [20..23], hg2 -> [24..27]
    for (int j = wave; j < 12; j += 8) {
      if (j < 4)      stage_frag(h1, 128, 128, j * 32,       pool + (size_t)(16 + j) * FR, lane);
      else if (j < 8) stage_frag(ga, 128, 128, (j - 4) * 32, pool + (size_t)(16 + j) * FR, lane);
      else            stage_frag(gv, 128, 128, (j - 8) * 32, pool + (size_t)(16 + j) * FR, lane);
    }
    __syncthreads();

    // S12: c_hat = tanh(...), g1 = sigmoid(...), g2 = sigmoid(...)
    for (int t = wave; t < 24; t += 8) {
      if (t < 8)       gemm_ct(pool + 16 * FR, wf + OFF_A2W2, 4, t,      lane, chat, 128, a2b2, nullptr, 3);
      else if (t < 16) gemm_ct(pool + 20 * FR, wf + OFF_G1W2, 4, t - 8,  lane, g1v, 128, g1b2, nullptr, 2);
      else             gemm_ct(pool + 24 * FR, wf + OFF_G2W2, 4, t - 16, lane, g2v, 128, g2b2, nullptr, 2);
    }
    __syncthreads();

    // S13: m = g1*m + g2*c_hat
    for (int idx = tid; idx < 16 * 128; idx += 256)
      mst[idx] = g1v[idx] * mst[idx] + g2v[idx] * chat[idx];
    __syncthreads();
  }

  // --- output head: last_h = [th, ah, vh, m] (320 cols) ---
  for (int idx = tid; idx < 16 * 320; idx += 256) {
    const int r = idx / 320, j = idx - r * 320;
    float v;
    if (j < 128)      v = th[r * 128 + j];
    else if (j < 160) v = ah[r * 32 + (j - 128)];
    else if (j < 192) v = vh[r * 32 + (j - 160)];
    else              v = mst[r * 128 + (j - 192)];
    cstar[idx] = v;   // reuse cstar as last_h[16][320]
  }
  __syncthreads();
  for (int j = wave; j < 10; j += 8) stage_frag(cstar, 320, 320, j * 32, pool + (size_t)j * FR, lane);
  __syncthreads();
  for (int t = wave; t < 8; t += 8) gemm_ct(pool, wf + OFF_OW1, 10, t, lane, h1, 128, ob1, nullptr, 1);
  __syncthreads();
  if (tid < 16) {
    float sacc = ob2[0];
    for (int j = 0; j < 128; ++j) sacc += h1[tid * 128 + j] * ow2[j];
    out[b0 + tid] = sacc;
  }
}

// ---------------------------------------------------------------------------
// Host entry
// ---------------------------------------------------------------------------
extern "C" void kernel_launch(void* const* d_in, const int* in_sizes, int n_in,
                              void* d_out, int out_size, void* d_ws, size_t ws_size,
                              hipStream_t stream) {
  (void)in_sizes; (void)n_in; (void)out_size; (void)ws_size;
  const float* x_p   = (const float*)d_in[0];
  const float* c_t0  = (const float*)d_in[1];
  const float* c_a0  = (const float*)d_in[2];
  const float* c_v0  = (const float*)d_in[3];
  const float* mem0  = (const float*)d_in[4];
  const float* t_Wih = (const float*)d_in[5];
  const float* t_Whh = (const float*)d_in[6];
  const float* t_bih = (const float*)d_in[7];
  const float* t_bhh = (const float*)d_in[8];
  const float* a_Wih = (const float*)d_in[9];
  const float* a_Whh = (const float*)d_in[10];
  const float* a_bih = (const float*)d_in[11];
  const float* a_bhh = (const float*)d_in[12];
  const float* v_Wih = (const float*)d_in[13];
  const float* v_Whh = (const float*)d_in[14];
  const float* v_bih = (const float*)d_in[15];
  const float* v_bhh = (const float*)d_in[16];
  const float* a1w1  = (const float*)d_in[17];
  const float* a1b1  = (const float*)d_in[18];
  const float* a1w2  = (const float*)d_in[19];
  const float* a1b2  = (const float*)d_in[20];
  const float* a2w1  = (const float*)d_in[21];
  const float* a2b1  = (const float*)d_in[22];
  const float* a2w2  = (const float*)d_in[23];
  const float* a2b2  = (const float*)d_in[24];
  const float* g1w1  = (const float*)d_in[25];
  const float* g1b1  = (const float*)d_in[26];
  const float* g1w2  = (const float*)d_in[27];
  const float* g1b2  = (const float*)d_in[28];
  const float* g2w1  = (const float*)d_in[29];
  const float* g2b1  = (const float*)d_in[30];
  const float* g2w2  = (const float*)d_in[31];
  const float* g2b2  = (const float*)d_in[32];
  const float* ow1   = (const float*)d_in[33];
  const float* ob1   = (const float*)d_in[34];
  const float* ow2   = (const float*)d_in[35];
  const float* ob2   = (const float*)d_in[36];

  _Float16* wf = (_Float16*)d_ws;

  // One-time (per launch, idempotent) weight swizzle to f16 WMMA fragments.
  auto prep = [&](const float* W1, int K1, int K1pad, const float* W2, int K2,
                  int N, size_t off) {
    const int ktiles = (K1pad + K2) / 32;
    const int total = (N / 16) * ktiles * 32;
    prep_weight_frags<<<(total + 255) / 256, 256, 0, stream>>>(W1, K1, K1pad, W2, K2, N,
                                                               wf + off);
  };
  prep(t_Wih, 300, 320, t_Whh, 128, 512, OFF_TCAT);
  prep(a_Wih,  81,  96, a_Whh,  32, 128, OFF_ACAT);
  prep(v_Wih, 371, 384, v_Whh,  32, 128, OFF_VCAT);
  prep(a1w1, 384, 384, nullptr, 0, 128, OFF_A1W1);
  prep(a1w2, 128, 128, nullptr, 0, 384, OFF_A1W2);
  prep(a2w1, 384, 384, nullptr, 0, 128, OFF_A2W1);
  prep(a2w2, 128, 128, nullptr, 0, 128, OFF_A2W2);
  prep(g1w1, 512, 512, nullptr, 0, 128, OFF_G1W1);
  prep(g1w2, 128, 128, nullptr, 0, 128, OFF_G1W2);
  prep(g2w1, 512, 512, nullptr, 0, 128, OFF_G2W1);
  prep(g2w2, 128, 128, nullptr, 0, 128, OFF_G2W2);
  prep(ow1,  320, 320, nullptr, 0, 128, OFF_OW1);

  // Persistent recurrent kernel: 32 blocks (B/16), 8 wave32 each, 192KB LDS.
  constexpr size_t LDS_BYTES = 196608;
  mfn_rnn_kernel<<<32, 256, LDS_BYTES, stream>>>(
      x_p, c_t0, c_a0, c_v0, mem0,
      t_bih, t_bhh, a_bih, a_bhh, v_bih, v_bhh,
      a1b1, a1b2, a2b1, a2b2,
      g1b1, g1b2, g2b1, g2b2,
      ob1, ow2, ob2,
      (const _Float16*)wf, (float*)d_out);
}
